// Temporal_Attention_Net_27522150433345
// MI455X (gfx1250) — matrix operand order repaired and checked by reading, not compiled, not run on hardware
//
#include <hip/hip_runtime.h>

// ---------------------------------------------------------------------------
// CDNA5 (gfx1250, wave32) implementation of the Temporal Attention Net.
// All GEMMs use v_wmma_f32_16x16x32_bf16 (f32 accumulate, bf16 operands).
// ---------------------------------------------------------------------------

typedef __attribute__((ext_vector_type(16))) __bf16 v16bf;
typedef __attribute__((ext_vector_type(8)))  float  v8f;

#define N_TOT 65408   // total flat timesteps
#define B_P   256     // persons
#define D_IN  512     // input size
#define H_SZ  512     // hidden size
#define G4H   2048    // 4*H
#define T_MAX 383     // max sequence length
#define ATT_BIAS 0.0001f

// ---------------------------- WMMA helpers ---------------------------------

static __device__ __forceinline__ v8f wmma_bf16(v16bf a, v16bf b, v8f c) {
  // 8 args: (neg_a, A, neg_b, B, c_mod, C, reuse_a, reuse_b)
  return __builtin_amdgcn_wmma_f32_16x16x32_bf16(false, a, false, b,
                                                 (short)0, c, false, false);
}

// A-fragment (16x32 bf16, M x K) from a row-major bf16 matrix, stride lda.
// ISA layout: lane = hi*16+m holds K = hi*8..hi*8+7 (halves 0..7) and
// K = 16+hi*8..16+hi*8+7 (halves 8..15) of row M=m.
static __device__ __forceinline__ v16bf load_a_bf(const __bf16* A, int lda,
                                                  int m0, int k0, int lane) {
  int m = lane & 15, hi = lane >> 4;
  const __bf16* p = A + (size_t)(m0 + m) * lda + k0 + hi * 8;
  v16bf a;
#pragma unroll
  for (int j = 0; j < 8; ++j) { a[j] = p[j]; a[8 + j] = p[16 + j]; }
  return a;
}

// A-fragment from a per-lane f32 row pointer (one row per lane's M), with a
// validity mask (invalid rows contribute zeros == reference zero padding).
static __device__ __forceinline__ v16bf load_a_f32_masked(const float* rowp,
                                                          bool valid, int k0,
                                                          int lane) {
  v16bf a;
  int hi = lane >> 4;
  if (valid) {
    const float* p = rowp + k0 + hi * 8;
#pragma unroll
    for (int j = 0; j < 8; ++j) {
      a[j]     = (__bf16)p[j];
      a[8 + j] = (__bf16)p[16 + j];
    }
  } else {
#pragma unroll
    for (int j = 0; j < 16; ++j) a[j] = (__bf16)0.0f;
  }
  return a;
}

// B-fragment (32x16 bf16, K x N) where B[k,n] = W[n,k] (W row-major, ld=ldw).
// ISA layout: lane = hi*16+n holds K = hi*8..hi*8+7 (halves 0..7) and
// K = 16+hi*8..16+hi*8+7 (halves 8..15) of column N=n.
static __device__ __forceinline__ v16bf load_b_w(const __bf16* W, int ldw,
                                                 int n0, int k0, int lane) {
  int n = lane & 15, kb = (lane >> 4) * 8;
  const __bf16* p = W + (size_t)(n0 + n) * ldw + k0 + kb;
  v16bf b;
#pragma unroll
  for (int j = 0; j < 8; ++j) { b[j] = p[j]; b[8 + j] = p[16 + j]; }
  return b;
}

static __device__ __forceinline__ float sigmoidf_(float x) {
  return 1.0f / (1.0f + expf(-x));
}

// ---------------------------- setup kernels --------------------------------

// offsets = exclusive prefix sum of lengths; bias = b_ih + b_hh
__global__ void prep_kernel(const int* __restrict__ lengths,
                            const float* __restrict__ b_ih,
                            const float* __restrict__ b_hh,
                            int* __restrict__ offsets,
                            float* __restrict__ bias) {
  if (threadIdx.x == 0) {
    int acc = 0;
    for (int b = 0; b < B_P; ++b) { offsets[b] = acc; acc += lengths[b]; }
  }
  for (int j = threadIdx.x; j < G4H; j += blockDim.x)
    bias[j] = b_ih[j] + b_hh[j];
}

__global__ void cvt_bf16_kernel(const float* __restrict__ src,
                                __bf16* __restrict__ dst, int n) {
  int i = blockIdx.x * blockDim.x + threadIdx.x;
  if (i < n) dst[i] = (__bf16)src[i];
}

__global__ void zero_state_kernel(float* __restrict__ c,
                                  __bf16* __restrict__ hbf) {
  int i = blockIdx.x * blockDim.x + threadIdx.x;
  if (i < B_P * H_SZ) { c[i] = 0.0f; hbf[i] = (__bf16)0.0f; }
}

// ---------------------------- LSTM step ------------------------------------

// gates[b, 0:2048] = h[b] @ W_hh^T + x_t[b] @ W_ih^T + (b_ih + b_hh)
// One wave computes a 16(M) x 64(N) strip. 512 waves = 64 blocks x 8 waves.
__global__ void gates_gemm_kernel(const float* __restrict__ x,
                                  const __bf16* __restrict__ Whh,
                                  const __bf16* __restrict__ Wih,
                                  const __bf16* __restrict__ hbf,
                                  const float* __restrict__ bias,
                                  const int* __restrict__ offsets,
                                  const int* __restrict__ lengths, int t,
                                  float* __restrict__ gates) {
  int lane = threadIdx.x & 31;
  int wave = blockIdx.x * (blockDim.x >> 5) + (threadIdx.x >> 5);
  int m0 = (wave >> 5) * 16;   // 16 row tiles
  int n0 = (wave & 31) * 64;   // 32 column strips of 64

  v8f acc[4];
#pragma unroll
  for (int s = 0; s < 4; ++s)
#pragma unroll
    for (int j = 0; j < 8; ++j) acc[s][j] = 0.0f;

  int m = lane & 15;
  int person = m0 + m;
  bool valid = (t < lengths[person]);
  const float* xrow = x + (size_t)(offsets[person] + t) * D_IN;

  // recurrent part: K over H
#pragma unroll 1
  for (int k0 = 0; k0 < H_SZ; k0 += 32) {
    v16bf a = load_a_bf(hbf, H_SZ, m0, k0, lane);
#pragma unroll
    for (int s = 0; s < 4; ++s) {
      v16bf b = load_b_w(Whh, H_SZ, n0 + 16 * s, k0, lane);
      acc[s] = wmma_bf16(a, b, acc[s]);
    }
  }
  // input part: K over D (masked rows == zero padding)
#pragma unroll 1
  for (int k0 = 0; k0 < D_IN; k0 += 32) {
    v16bf a = load_a_f32_masked(xrow, valid, k0, lane);
#pragma unroll
    for (int s = 0; s < 4; ++s) {
      v16bf b = load_b_w(Wih, D_IN, n0 + 16 * s, k0, lane);
      acc[s] = wmma_bf16(a, b, acc[s]);
    }
  }

  int hi = lane >> 4;
#pragma unroll
  for (int s = 0; s < 4; ++s) {
    int col = n0 + 16 * s + m;     // m doubles as the N index for C/D layout
    float bc = bias[col];
#pragma unroll
    for (int r = 0; r < 8; ++r) {
      int row = m0 + r + 8 * hi;
      gates[(size_t)row * G4H + col] = acc[s][r] + bc;
    }
  }
}

// Pointwise LSTM cell: gate order i,f,g,o. Skips rows past their length
// (padding is at segment ends, so valid outputs match the reference).
__global__ void lstm_point_kernel(const float* __restrict__ gates,
                                  const int* __restrict__ offsets,
                                  const int* __restrict__ lengths, int t,
                                  float* __restrict__ c,
                                  __bf16* __restrict__ hbf,
                                  float* __restrict__ flat_f,
                                  __bf16* __restrict__ flat_bf) {
  int idx = blockIdx.x * blockDim.x + threadIdx.x;  // 0 .. B*H-1
  if (idx >= B_P * H_SZ) return;
  int b = idx >> 9;        // /512
  int j = idx & 511;
  if (t >= lengths[b]) return;

  const float* g = gates + (size_t)b * G4H;
  float ig = g[j];
  float fg = g[512 + j];
  float gg = g[1024 + j];
  float og = g[1536 + j];

  float cn = sigmoidf_(fg) * c[idx] + sigmoidf_(ig) * tanhf(gg);
  float hn = sigmoidf_(og) * tanhf(cn);
  c[idx] = cn;
  hbf[idx] = (__bf16)hn;

  size_t fo = ((size_t)(offsets[b] + t)) * H_SZ + j;
  flat_f[fo] = hn;
  flat_bf[fo] = (__bf16)hn;
}

// ---------------------------- attention head -------------------------------

// passed_Z = flat @ fc1_W^T + fc1_b   -> written straight into d_out
__global__ void fc1_gemm_kernel(const __bf16* __restrict__ flatbf,
                                const __bf16* __restrict__ fc1Wbf,
                                const float* __restrict__ fc1b,
                                float* __restrict__ passZ) {
  int lane = threadIdx.x & 31;
  int m0 = blockIdx.x * 16;                 // 4088 row tiles
  int n0 = (threadIdx.x >> 5) * 64;         // 8 waves -> 512 cols

  v8f acc[4];
#pragma unroll
  for (int s = 0; s < 4; ++s)
#pragma unroll
    for (int j = 0; j < 8; ++j) acc[s][j] = 0.0f;

#pragma unroll 1
  for (int k0 = 0; k0 < H_SZ; k0 += 32) {
    v16bf a = load_a_bf(flatbf, H_SZ, m0, k0, lane);
#pragma unroll
    for (int s = 0; s < 4; ++s) {
      v16bf b = load_b_w(fc1Wbf, H_SZ, n0 + 16 * s, k0, lane);
      acc[s] = wmma_bf16(a, b, acc[s]);
    }
  }

  int m = lane & 15, hi = lane >> 4;
#pragma unroll
  for (int s = 0; s < 4; ++s) {
    int col = n0 + 16 * s + m;
    float bc = fc1b[col];
#pragma unroll
    for (int r = 0; r < 8; ++r) {
      int row = m0 + r + 8 * hi;
      passZ[(size_t)row * D_IN + col] = acc[s][r] + bc;
    }
  }
}

// att[n] = sigmoid(log(eps) - log(1-eps) + passed_Z[n] . fc2_W + fc2_b)
// one wave per row, shuffle reduction.
__global__ void fc2_att_kernel(const float* __restrict__ passZ,
                               const float* __restrict__ fc2W,
                               const float* __restrict__ fc2b,
                               const float* __restrict__ eps_u,
                               float* __restrict__ att) {
  int lane = threadIdx.x & 31;
  int row = blockIdx.x * (blockDim.x >> 5) + (threadIdx.x >> 5);
  if (row >= N_TOT) return;
  const float* pz = passZ + (size_t)row * D_IN;
  float s = 0.0f;
#pragma unroll 4
  for (int j = lane; j < D_IN; j += 32) s += pz[j] * fc2W[j];
#pragma unroll
  for (int off = 16; off > 0; off >>= 1) s += __shfl_down(s, off, 32);
  if (lane == 0) {
    float logit = s + fc2b[0];
    float eps = (2.0f * ATT_BIAS - 1.0f) * eps_u[row] + (1.0f - ATT_BIAS);
    float g = logf(eps) - logf(1.0f - eps) + logit;
    att[row] = sigmoidf_(g);
  }
}

// Deterministic per-person segment sum: seg[b] = sum_p flat[off+p] * att[off+p]
__global__ void seg_sum_kernel(const float* __restrict__ flat_f,
                               const float* __restrict__ att,
                               const int* __restrict__ offsets,
                               const int* __restrict__ lengths,
                               float* __restrict__ seg) {
  int b = blockIdx.x;
  int j = threadIdx.x;  // 256 threads, each handles cols j and j+256
  int off = offsets[b], len = lengths[b];
  float s0 = 0.0f, s1 = 0.0f;
  for (int p = 0; p < len; ++p) {
    size_t n = (size_t)(off + p);
    float a = att[n];
    const float* f = flat_f + n * H_SZ;
    s0 += f[j] * a;
    s1 += f[256 + j] * a;
  }
  seg[(size_t)b * H_SZ + j] = s0;
  seg[(size_t)b * H_SZ + 256 + j] = s1;
}

// final = (seg @ mlp1_W^T + mlp1_b) @ mlp2_W^T + mlp2_b   (tiny: scalar)
__global__ void mlp_kernel(const float* __restrict__ seg,
                           const float* __restrict__ m1W,
                           const float* __restrict__ m1b,
                           const float* __restrict__ m2W,
                           const float* __restrict__ m2b,
                           float* __restrict__ outp) {
  int b = threadIdx.x;  // 256 threads, one person each
  float hid[16];
  const float* sg = seg + (size_t)b * H_SZ;
#pragma unroll
  for (int k = 0; k < 16; ++k) {
    float s = m1b[k];
    const float* w = m1W + (size_t)k * H_SZ;
    for (int h = 0; h < H_SZ; ++h) s += sg[h] * w[h];
    hid[k] = s;
  }
#pragma unroll
  for (int o = 0; o < 2; ++o) {
    float s = m2b[o];
#pragma unroll
    for (int k = 0; k < 16; ++k) s += hid[k] * m2W[o * 16 + k];
    outp[b * 2 + o] = s;
  }
}

// ---------------------------- host driver ----------------------------------

static inline size_t align256(size_t x) { return (x + 255) & ~(size_t)255; }

extern "C" void kernel_launch(void* const* d_in, const int* in_sizes, int n_in,
                              void* d_out, int out_size, void* d_ws,
                              size_t ws_size, hipStream_t stream) {
  const float* lstm_input = (const float*)d_in[0];
  const float* eps_u      = (const float*)d_in[1];
  const float* W_ih       = (const float*)d_in[2];
  const float* W_hh       = (const float*)d_in[3];
  const float* b_ih       = (const float*)d_in[4];
  const float* b_hh       = (const float*)d_in[5];
  const float* fc1_W      = (const float*)d_in[6];
  const float* fc1_b      = (const float*)d_in[7];
  const float* fc2_W      = (const float*)d_in[8];
  const float* fc2_b      = (const float*)d_in[9];
  const float* mlp1_W     = (const float*)d_in[10];
  const float* mlp1_b     = (const float*)d_in[11];
  const float* mlp2_W     = (const float*)d_in[12];
  const float* mlp2_b     = (const float*)d_in[13];
  const int*   lengths    = (const int*)d_in[14];

  float* out_final = (float*)d_out;            // [B, 2]
  float* out_passZ = (float*)d_out + B_P * 2;  // [N, D]

  // workspace carve-up
  char* ws = (char*)d_ws;
  size_t off = 0;
  int*    offsets = (int*)(ws + off);    off = align256(off + B_P * 4);
  float*  bias    = (float*)(ws + off);  off = align256(off + G4H * 4);
  __bf16* WhhBf   = (__bf16*)(ws + off); off = align256(off + (size_t)G4H * H_SZ * 2);
  __bf16* WihBf   = (__bf16*)(ws + off); off = align256(off + (size_t)G4H * D_IN * 2);
  __bf16* fc1WBf  = (__bf16*)(ws + off); off = align256(off + (size_t)D_IN * H_SZ * 2);
  __bf16* hBf     = (__bf16*)(ws + off); off = align256(off + (size_t)B_P * H_SZ * 2);
  float*  cState  = (float*)(ws + off);  off = align256(off + (size_t)B_P * H_SZ * 4);
  float*  gates   = (float*)(ws + off);  off = align256(off + (size_t)B_P * G4H * 4);
  float*  att     = (float*)(ws + off);  off = align256(off + (size_t)N_TOT * 4);
  float*  seg     = (float*)(ws + off);  off = align256(off + (size_t)B_P * H_SZ * 4);
  float*  flatF   = (float*)(ws + off);  off = align256(off + (size_t)N_TOT * H_SZ * 4);
  __bf16* flatBf  = (__bf16*)(ws + off); off = align256(off + (size_t)N_TOT * H_SZ * 2);
  (void)ws_size; (void)in_sizes; (void)n_in; (void)out_size;

  // setup
  prep_kernel<<<1, 256, 0, stream>>>(lengths, b_ih, b_hh, offsets, bias);
  cvt_bf16_kernel<<<(G4H * H_SZ + 255) / 256, 256, 0, stream>>>(W_hh, WhhBf, G4H * H_SZ);
  cvt_bf16_kernel<<<(G4H * D_IN + 255) / 256, 256, 0, stream>>>(W_ih, WihBf, G4H * D_IN);
  cvt_bf16_kernel<<<(D_IN * H_SZ + 255) / 256, 256, 0, stream>>>(fc1_W, fc1WBf, D_IN * H_SZ);
  zero_state_kernel<<<(B_P * H_SZ + 255) / 256, 256, 0, stream>>>(cState, hBf);

  // sequential LSTM scan
  for (int t = 0; t < T_MAX; ++t) {
    gates_gemm_kernel<<<64, 256, 0, stream>>>(lstm_input, WhhBf, WihBf, hBf,
                                              bias, offsets, lengths, t, gates);
    lstm_point_kernel<<<(B_P * H_SZ + 255) / 256, 256, 0, stream>>>(
        gates, offsets, lengths, t, cState, hBf, flatF, flatBf);
  }

  // attention head
  fc1_gemm_kernel<<<N_TOT / 16, 256, 0, stream>>>(flatBf, fc1WBf, fc1_b, out_passZ);
  fc2_att_kernel<<<N_TOT / 8, 256, 0, stream>>>(out_passZ, fc2_W, fc2_b, eps_u, att);
  seg_sum_kernel<<<B_P, 256, 0, stream>>>(flatF, att, offsets, lengths, seg);
  mlp_kernel<<<1, 256, 0, stream>>>(seg, mlp1_W, mlp1_b, mlp2_W, mlp2_b, out_final);
}
